// Block_80977313399440
// MI455X (gfx1250) — compile-verified, matrix operand order checked by reading, not run on hardware
//
#include <hip/hip_runtime.h>
#include <hip/hip_bf16.h>

// ---------------------------------------------------------------------------
// Types for CDNA5 WMMA (wave32): bf16 A/B fragments (16 x bf16 per lane),
// f32 C/D accumulator (8 x f32 per lane).
// ---------------------------------------------------------------------------
typedef __attribute__((ext_vector_type(16))) __bf16 v16bf;
typedef __attribute__((ext_vector_type(8)))  float  v8f;
typedef __attribute__((ext_vector_type(2)))  unsigned int u32x2;
typedef __attribute__((ext_vector_type(4)))  unsigned int u32x4;

#define TOKENS 8192      // B*N = 8*1024
#define D_EMB  1024
#define D_FF   4096

// ---------------------------------------------------------------------------
// CDNA5 async global->LDS path (ASYNCcnt), guarded so compilation never breaks
// ---------------------------------------------------------------------------
#define GLOBAL_AS __attribute__((address_space(1)))
#define LDS_AS    __attribute__((address_space(3)))

// GCC-style vector matching the builtin prototype:
//   __builtin_amdgcn_global_load_async_to_lds_b64(v2i AS1*, v2i AS3*, Ii, Ii)
typedef int gcc_i32x2 __attribute__((vector_size(8)));

#if defined(__gfx1250__) && \
    __has_builtin(__builtin_amdgcn_global_load_async_to_lds_b64) && \
    __has_builtin(__builtin_amdgcn_s_wait_asynccnt)
#define HAVE_ASYNC_LDS 1
#else
#define HAVE_ASYNC_LDS 0
#endif

// Copy 8 bytes (4 bf16) global -> LDS, per lane.
__device__ __forceinline__ void copy8_g2l(const __bf16* g, __bf16* l) {
#if HAVE_ASYNC_LDS
    __builtin_amdgcn_global_load_async_to_lds_b64(
        (GLOBAL_AS gcc_i32x2*)(const void*)g,
        (LDS_AS    gcc_i32x2*)(void*)l, 0, 0);
#else
    *reinterpret_cast<u32x2*>(l) = *reinterpret_cast<const u32x2*>(g);
#endif
}
__device__ __forceinline__ void wait_async_g2l() {
#if HAVE_ASYNC_LDS
    __builtin_amdgcn_s_wait_asynccnt(0);
#endif
}

// ---------------------------------------------------------------------------
// WMMA helpers
// ---------------------------------------------------------------------------
__device__ __forceinline__ v8f wmma_bf16(v16bf a, v16bf b, v8f c) {
    return __builtin_amdgcn_wmma_f32_16x16x32_bf16(
        false, a, false, b, (short)0, c, false, false);
}

// Load one 16x32 bf16 fragment from LDS (ISA 7.12.2 16-bit layout):
//   lanes 0-15 : row = rowBase+lane,    K-runs {0..7, 16..23}
//   lanes 16-31: row = rowBase+lane-16, K-runs {8..15, 24..31}
__device__ __forceinline__ v16bf load_frag(const __bf16* s, int stride,
                                           int rowBase, int colBase, int lane) {
    int lr  = lane & 15;
    int sel = (lane >> 4) & 1;
    const __bf16* p = s + (size_t)(rowBase + lr) * stride + colBase + sel * 8;
    union { u32x4 q[2]; v16bf v; } u;
    u.q[0] = *reinterpret_cast<const u32x4*>(p);
    u.q[1] = *reinterpret_cast<const u32x4*>(p + 16);
    return u.v;
}

// ---------------------------------------------------------------------------
// Weight conversion kernels (f32 -> bf16)
// ---------------------------------------------------------------------------
__global__ __launch_bounds__(256) void conv_flat_kernel(const float* __restrict__ in,
                                                        __bf16* __restrict__ out, int n) {
    int i = blockIdx.x * 256 + threadIdx.x;
    if (i < n) out[i] = (__bf16)in[i];
}

// Wq/Wk/Wv: [H, D, HS] -> row-major [D, H*HS] bf16
__global__ __launch_bounds__(256) void conv_qkvw_kernel(const float* __restrict__ W,
                                                        __bf16* __restrict__ out) {
    int i = blockIdx.x * 256 + threadIdx.x;          // i < 1024*1024
    int d = i >> 10;
    int col = i & 1023;
    int h = col >> 6;
    int s = col & 63;
    out[i] = (__bf16)W[((size_t)h << 16) + ((size_t)d << 6) + s];
}

// ---------------------------------------------------------------------------
// LayerNorm: f32 [rows, 1024] -> bf16 [rows, 1024]
// ---------------------------------------------------------------------------
__global__ __launch_bounds__(256) void ln_kernel(const float* __restrict__ x,
                                                 const float* __restrict__ g,
                                                 const float* __restrict__ bta,
                                                 __bf16* __restrict__ out) {
    __shared__ float s1[256];
    __shared__ float s2[256];
    const int row = blockIdx.x;
    const int tid = threadIdx.x;
    const float* xr = x + (size_t)row * D_EMB;

    float sum = 0.f, sq = 0.f;
    float v[4];
#pragma unroll
    for (int i = 0; i < 4; ++i) {
        v[i] = xr[tid + i * 256];
        sum += v[i];
        sq  += v[i] * v[i];
    }
    s1[tid] = sum; s2[tid] = sq;
    __syncthreads();
    for (int off = 128; off > 0; off >>= 1) {
        if (tid < off) { s1[tid] += s1[tid + off]; s2[tid] += s2[tid + off]; }
        __syncthreads();
    }
    const float mean = s1[0] * (1.f / D_EMB);
    const float var  = s2[0] * (1.f / D_EMB) - mean * mean;
    const float rstd = rsqrtf(var + 1e-5f);

    __bf16* orow = out + (size_t)row * D_EMB;
#pragma unroll
    for (int i = 0; i < 4; ++i) {
        int idx = tid + i * 256;
        float y = (v[i] - mean) * rstd * g[idx] + bta[idx];
        orow[idx] = (__bf16)y;
    }
}

// ---------------------------------------------------------------------------
// Tiled bf16 GEMM, C[M,N] = A[M,K]*B[K,N] + bias, templated fused epilogue:
//   MODE 0: out_bf16 = acc + bias
//   MODE 1: out_f32  = acc + bias + resid
//   MODE 2: out_bf16 = leakyrelu(acc + bias, 0.01)
// 256 threads (8 waves). Block tile 128x128, BK=64, double-buffered LDS.
// Wave tile 32(m) x 64(n): 8 accumulators; per K-step 2 A-frags + 4 B-frags
// feed 8 WMMAs (16 WMMAs per 64-K slab, one barrier per slab).
// ---------------------------------------------------------------------------
template <int MODE>
__global__ __launch_bounds__(256) void gemm_bf16_kernel(
    const __bf16* __restrict__ A, const __bf16* __restrict__ B,
    const float* __restrict__ bias, const float* __restrict__ resid,
    __bf16* __restrict__ outb, float* __restrict__ outf,
    int M, int N, int K) {

    __shared__ __align__(16) __bf16 As[2][128 * 64];   // [m][k], stride 64
    __shared__ __align__(16) __bf16 Bs[2][128 * 64];   // [n][k] (transposed)

    const int tid  = threadIdx.x;
    const int lane = tid & 31;
    const int wave = tid >> 5;
    const int wm = wave >> 1;       // 0..3 : 32-row slice
    const int wn = wave & 1;        // 0..1 : 64-col slice
    const int m0 = blockIdx.y * 128;
    const int n0 = blockIdx.x * 128;

    auto load_tiles = [&](int p, int k0) {
        // A tile 128x64: contiguous per-lane 8B -> async to LDS
#pragma unroll
        for (int i = 0; i < 8; ++i) {
            int e  = tid + i * 256;          // 2048 vec4 slots
            int r  = e >> 4;
            int c4 = (e & 15) * 4;
            copy8_g2l(&A[(size_t)(m0 + r) * K + k0 + c4], &As[p][r * 64 + c4]);
        }
        // B tile 64x128, transposed into [n][k] (scatter: sync path)
#pragma unroll
        for (int i = 0; i < 8; ++i) {
            int e  = tid + i * 256;
            int kk = e >> 5;                 // 0..63
            int n4 = (e & 31) * 4;           // 0..124
            union { u32x2 d; __bf16 h[4]; } u;
            u.d = *reinterpret_cast<const u32x2*>(&B[(size_t)(k0 + kk) * N + n0 + n4]);
#pragma unroll
            for (int j = 0; j < 4; ++j) Bs[p][(n4 + j) * 64 + kk] = u.h[j];
        }
    };

    const v8f vzero = {0.f,0.f,0.f,0.f,0.f,0.f,0.f,0.f};
    v8f acc[2][4];
#pragma unroll
    for (int mt = 0; mt < 2; ++mt)
#pragma unroll
        for (int nt = 0; nt < 4; ++nt) acc[mt][nt] = vzero;

    load_tiles(0, 0);
    wait_async_g2l();
    __syncthreads();

    int p = 0;
    for (int k0 = 0; k0 < K; k0 += 64) {
        if (k0 + 64 < K) load_tiles(p ^ 1, k0 + 64);   // prefetch next slab

#pragma unroll
        for (int ks = 0; ks < 2; ++ks) {
            v16bf a0 = load_frag(As[p], 64, wm * 32 +  0, ks * 32, lane);
            v16bf a1 = load_frag(As[p], 64, wm * 32 + 16, ks * 32, lane);
#pragma unroll
            for (int nt = 0; nt < 4; ++nt) {
                v16bf b = load_frag(Bs[p], 64, wn * 64 + nt * 16, ks * 32, lane);
                acc[0][nt] = wmma_bf16(a0, b, acc[0][nt]);
                acc[1][nt] = wmma_bf16(a1, b, acc[1][nt]);
            }
        }
        wait_async_g2l();
        __syncthreads();
        p ^= 1;
    }

    // ---- straight-line epilogue (MODE is a compile-time constant) ----
    const int lr = lane & 15;
    const int hi = lane >> 4;
#pragma unroll
    for (int mt = 0; mt < 2; ++mt) {
#pragma unroll
        for (int nt = 0; nt < 4; ++nt) {
#pragma unroll
            for (int j = 0; j < 8; ++j) {
                int row = m0 + wm * 32 + mt * 16 + j + hi * 8;
                int col = n0 + wn * 64 + nt * 16 + lr;
                float v = acc[mt][nt][j] + bias[col];
                size_t idx = (size_t)row * N + col;
                if (MODE == 2) v = (v > 0.f) ? v : 0.01f * v;
                if (MODE == 1) outf[idx] = v + resid[idx];
                else           outb[idx] = (__bf16)v;
            }
        }
    }
}

// ---------------------------------------------------------------------------
// Flash attention: one workgroup per (b, h, 64-query block).
// q/k/v/o are bf16 [TOKENS, 1024] token-major; head h = cols [h*64, h*64+64).
// 128 threads = 4 waves; wave w owns query rows [16w, 16w+16).
// ---------------------------------------------------------------------------
__global__ __launch_bounds__(128) void attn_kernel(
    const __bf16* __restrict__ q, const __bf16* __restrict__ k,
    const __bf16* __restrict__ v, __bf16* __restrict__ o) {

    __shared__ __align__(16) __bf16 Qs[64 * 64];   // [query][hs]
    __shared__ __align__(16) __bf16 Ks[64 * 64];   // [key][hs]
    __shared__ __align__(16) __bf16 Vs[64 * 64];   // [hs][key] (transposed)
    __shared__ __align__(16) __bf16 Ps[64 * 64];   // softmax probs
    __shared__ __align__(16) float  Ss[64 * 64];   // scores f32
    __shared__ float row_m[64], row_l[64], row_scale[64];

    const int tid  = threadIdx.x;
    const int lane = tid & 31;
    const int wave = tid >> 5;
    const int lr = lane & 15;
    const int hi = lane >> 4;

    const int bid = blockIdx.x;           // 2048 blocks
    const int qb  = bid & 15;
    const int h   = (bid >> 4) & 15;
    const int b   = bid >> 8;
    const int q0  = qb * 64;
    const size_t tokBase = (size_t)b * 1024;
    const size_t colBase = (size_t)h * 64;

    // load Q block [64 x 64] (async path when available)
#pragma unroll
    for (int i = 0; i < 8; ++i) {
        int e  = tid + i * 128;
        int r  = e >> 4;
        int s4 = (e & 15) * 4;
        copy8_g2l(&q[(tokBase + q0 + r) * D_EMB + colBase + s4], &Qs[r * 64 + s4]);
    }
    if (tid < 64) { row_m[tid] = -3e38f; row_l[tid] = 0.f; }

    const v8f vzero = {0.f,0.f,0.f,0.f,0.f,0.f,0.f,0.f};
    v8f acc_o[4];
#pragma unroll
    for (int c = 0; c < 4; ++c) acc_o[c] = vzero;

    const int rowBase = wave * 16;

    for (int kb = 0; kb < 16; ++kb) {
        const int kt0 = kb * 64;
#pragma unroll
        for (int i = 0; i < 8; ++i) {
            int e  = tid + i * 128;
            int r  = e >> 4;
            int s4 = (e & 15) * 4;
            copy8_g2l(&k[(tokBase + kt0 + r) * D_EMB + colBase + s4], &Ks[r * 64 + s4]);
            union { u32x2 d; __bf16 hh[4]; } u;
            u.d = *reinterpret_cast<const u32x2*>(&v[(tokBase + kt0 + r) * D_EMB + colBase + s4]);
#pragma unroll
            for (int j = 0; j < 4; ++j) Vs[(s4 + j) * 64 + r] = u.hh[j];
        }
        wait_async_g2l();
        __syncthreads();

        // S = (Q K^T) * 1/sqrt(64)
        v16bf a0 = load_frag(Qs, 64, rowBase, 0,  lane);
        v16bf a1 = load_frag(Qs, 64, rowBase, 32, lane);
#pragma unroll
        for (int c = 0; c < 4; ++c) {
            v16bf bk0 = load_frag(Ks, 64, c * 16, 0,  lane);  // lane = key col
            v16bf bk1 = load_frag(Ks, 64, c * 16, 32, lane);
            v8f s = wmma_bf16(a0, bk0, vzero);
            s     = wmma_bf16(a1, bk1, s);
#pragma unroll
            for (int j = 0; j < 8; ++j)
                Ss[(rowBase + j + hi * 8) * 64 + c * 16 + lr] = s[j] * 0.125f;
        }
        __syncthreads();

        // online softmax per row
        if (tid < 64) {
            const int r = tid;
            float bm = -3e38f;
#pragma unroll 8
            for (int j = 0; j < 64; ++j) bm = fmaxf(bm, Ss[r * 64 + j]);
            float nm = fmaxf(row_m[r], bm);
            float sc = __expf(row_m[r] - nm);
            float sum = 0.f;
#pragma unroll 8
            for (int j = 0; j < 64; ++j) {
                float p = __expf(Ss[r * 64 + j] - nm);
                Ps[r * 64 + j] = (__bf16)p;
                sum += p;
            }
            row_l[r] = row_l[r] * sc + sum;
            row_m[r] = nm;
            row_scale[r] = sc;
        }
        __syncthreads();

        // rescale running O, then O += P * V
        float f[8];
#pragma unroll
        for (int j = 0; j < 8; ++j) f[j] = row_scale[rowBase + j + hi * 8];
#pragma unroll
        for (int c = 0; c < 4; ++c)
#pragma unroll
            for (int j = 0; j < 8; ++j) acc_o[c][j] *= f[j];

        v16bf p0 = load_frag(Ps, 64, rowBase, 0,  lane);
        v16bf p1 = load_frag(Ps, 64, rowBase, 32, lane);
#pragma unroll
        for (int c = 0; c < 4; ++c) {
            v16bf bv0 = load_frag(Vs, 64, c * 16, 0,  lane);  // lane = hs col
            v16bf bv1 = load_frag(Vs, 64, c * 16, 32, lane);
            acc_o[c] = wmma_bf16(p0, bv0, acc_o[c]);
            acc_o[c] = wmma_bf16(p1, bv1, acc_o[c]);
        }
        __syncthreads();
    }

    // normalize and store O
    float il[8];
#pragma unroll
    for (int j = 0; j < 8; ++j) il[j] = 1.f / row_l[rowBase + j + hi * 8];
#pragma unroll
    for (int c = 0; c < 4; ++c)
#pragma unroll
        for (int j = 0; j < 8; ++j) {
            int r = rowBase + j + hi * 8;
            o[(tokBase + q0 + r) * D_EMB + colBase + c * 16 + lr] =
                (__bf16)(acc_o[c][j] * il[j]);
        }
}

// ---------------------------------------------------------------------------
// Host launch
// ---------------------------------------------------------------------------
extern "C" void kernel_launch(void* const* d_in, const int* in_sizes, int n_in,
                              void* d_out, int out_size, void* d_ws, size_t ws_size,
                              hipStream_t stream) {
    const float* x      = (const float*)d_in[0];
    const float* Wq     = (const float*)d_in[1];
    const float* bq     = (const float*)d_in[2];
    const float* Wk     = (const float*)d_in[3];
    const float* bk     = (const float*)d_in[4];
    const float* Wv     = (const float*)d_in[5];
    const float* bv     = (const float*)d_in[6];
    const float* Wproj  = (const float*)d_in[7];
    const float* bproj  = (const float*)d_in[8];
    const float* W1     = (const float*)d_in[9];
    const float* b1     = (const float*)d_in[10];
    const float* W2     = (const float*)d_in[11];
    const float* b2     = (const float*)d_in[12];
    const float* ln1_g  = (const float*)d_in[13];
    const float* ln1_b  = (const float*)d_in[14];
    const float* ln2_g  = (const float*)d_in[15];
    const float* ln2_b  = (const float*)d_in[16];
    float* out = (float*)d_out;

    const size_t MB = 1u << 20;
    char* ws = (char*)d_ws;
    __bf16* Wqb    = (__bf16*)(ws +   0 * MB);   //  2 MB
    __bf16* Wkb    = (__bf16*)(ws +   2 * MB);   //  2 MB
    __bf16* Wvb    = (__bf16*)(ws +   4 * MB);   //  2 MB
    __bf16* Wprojb = (__bf16*)(ws +   6 * MB);   //  2 MB
    __bf16* W1b    = (__bf16*)(ws +   8 * MB);   //  8 MB
    __bf16* W2b    = (__bf16*)(ws +  16 * MB);   //  8 MB
    __bf16* h_bf   = (__bf16*)(ws +  24 * MB);   // 16 MB  LN1 output
    __bf16* qbuf   = (__bf16*)(ws +  40 * MB);   // 16 MB
    __bf16* kbuf   = (__bf16*)(ws +  56 * MB);   // 16 MB
    __bf16* vbuf   = (__bf16*)(ws +  72 * MB);   // 16 MB
    __bf16* attnb  = (__bf16*)(ws +  88 * MB);   // 16 MB
    float*  x1     = (float*) (ws + 104 * MB);   // 32 MB  post-attn residual
    __bf16* h2_bf  = (__bf16*)(ws + 136 * MB);   // 16 MB  LN2 output
    __bf16* ffb    = (__bf16*)(ws + 152 * MB);   // 64 MB  FFN hidden

    // 1) weight conversions
    conv_qkvw_kernel<<<4096, 256, 0, stream>>>(Wq, Wqb);
    conv_qkvw_kernel<<<4096, 256, 0, stream>>>(Wk, Wkb);
    conv_qkvw_kernel<<<4096, 256, 0, stream>>>(Wv, Wvb);
    conv_flat_kernel<<<4096, 256, 0, stream>>>(Wproj, Wprojb, 1 << 20);
    conv_flat_kernel<<<16384, 256, 0, stream>>>(W1, W1b, 1 << 22);
    conv_flat_kernel<<<16384, 256, 0, stream>>>(W2, W2b, 1 << 22);

    // 2) LN1
    ln_kernel<<<TOKENS, 256, 0, stream>>>(x, ln1_g, ln1_b, h_bf);

    // 3) QKV projections
    dim3 gD(D_EMB / 128, TOKENS / 128);    // (8, 64)
    gemm_bf16_kernel<0><<<gD, 256, 0, stream>>>(h_bf, Wqb, bq, nullptr, qbuf, nullptr,
                                                TOKENS, D_EMB, D_EMB);
    gemm_bf16_kernel<0><<<gD, 256, 0, stream>>>(h_bf, Wkb, bk, nullptr, kbuf, nullptr,
                                                TOKENS, D_EMB, D_EMB);
    gemm_bf16_kernel<0><<<gD, 256, 0, stream>>>(h_bf, Wvb, bv, nullptr, vbuf, nullptr,
                                                TOKENS, D_EMB, D_EMB);

    // 4) attention
    attn_kernel<<<8 * 16 * 16, 128, 0, stream>>>(qbuf, kbuf, vbuf, attnb);

    // 5) output projection + residual (f32)
    gemm_bf16_kernel<1><<<gD, 256, 0, stream>>>(attnb, Wprojb, bproj, x, nullptr, x1,
                                                TOKENS, D_EMB, D_EMB);

    // 6) LN2
    ln_kernel<<<TOKENS, 256, 0, stream>>>(x1, ln2_g, ln2_b, h2_bf);

    // 7) FFN1 + LeakyReLU
    dim3 gFF(D_FF / 128, TOKENS / 128);    // (32, 64)
    gemm_bf16_kernel<2><<<gFF, 256, 0, stream>>>(h2_bf, W1b, b1, nullptr, ffb, nullptr,
                                                 TOKENS, D_FF, D_EMB);

    // 8) FFN2 + residual -> final f32 output
    gemm_bf16_kernel<1><<<gD, 256, 0, stream>>>(ffb, W2b, b2, x1, nullptr, out,
                                                TOKENS, D_EMB, D_FF);
}
